// LinOSSLayer_37735582663400
// MI455X (gfx1250) — compile-verified, matrix operand order checked by reading, not run on hardware
//
#include <hip/hip_runtime.h>
#include <hip/hip_bf16.h>

#define BB   16
#define HH   512
#define LL   4096
#define PP   64
#define CLEN 256            // scan chunk length (2^8)
#define CCH  (LL / CLEN)    // 16 chunks
#define WPAD 520            // LDS row stride in bf16 (1040B = 65*16B, 4-bank skew per row)

typedef __attribute__((ext_vector_type(16))) __bf16 v16bf;
typedef __attribute__((ext_vector_type(8)))  __bf16 v8bf;
typedef __attribute__((ext_vector_type(8)))  float  v8f;

// Pack a WMMA 16x16x32 bf16 fragment from a contiguous 32-element row slice.
// Element j (0..7)  -> k = 8*khalf + j
// Element j (8..15) -> k = 16 + 8*khalf + (j-8)
static __device__ __forceinline__ v16bf load_bf_frag(const __bf16* rowptr, int khalf) {
  v8bf lo = *(const v8bf*)(rowptr + (khalf << 3));
  v8bf hi = *(const v8bf*)(rowptr + 16 + (khalf << 3));
  v16bf f;
#pragma unroll
  for (int j = 0; j < 8; ++j) { f[j] = lo[j]; f[8 + j] = hi[j]; }
  return f;
}

// ---------------- per-channel scalars ----------------
__global__ void k_prep(const float* __restrict__ A_diag, const float* __restrict__ steps_raw,
                       float* __restrict__ scal) {
  int p = threadIdx.x;                       // 64 threads
  float A   = fmaxf(A_diag[p], 0.0f);
  float s   = 1.0f / (1.0f + expf(-steps_raw[p]));
  float sch = 1.0f / (1.0f + s * s * A);
  scal[p]          = 1.0f - s * s * A * sch; // M11
  scal[PP + p]     = -s * A * sch;           // M12
  scal[2 * PP + p] = s * sch;                // M21
  scal[3 * PP + p] = sch;                    // M22
  scal[4 * PP + p] = s;                      // steps
}

// ---------------- f32 -> bf16 (strided gather for Bp[...,0]/Cp[...,0]) ----------------
__global__ void k_cvt(const float* __restrict__ src, __bf16* __restrict__ dst, int n, int stride) {
  int i = blockIdx.x * blockDim.x + threadIdx.x;
  if (i < n) dst[i] = (__bf16)src[(size_t)i * stride];
}

// ---------------- GEMM 1: Bu[b,l,p] = sum_h Bp0[p,h] * u[b,h,l] ----------------
__global__ void __launch_bounds__(256) k_gemm_bu(const float* __restrict__ u,
                                                 const __bf16* __restrict__ Bp0,
                                                 float* __restrict__ Bu) {
  const int lane  = threadIdx.x & 31;
  const int wave  = threadIdx.x >> 5;
  const int tile  = blockIdx.x * 8 + wave;       // l-tile id over B*L/16
  const int b     = tile / (LL / 16);
  const int l0    = (tile % (LL / 16)) * 16;
  const int col   = lane & 15;
  const int khalf = lane >> 4;
  const float* ub = u + (size_t)b * HH * LL;

  v8f acc[4] = {{}, {}, {}, {}};
  for (int k0 = 0; k0 < HH; k0 += 32) {
    v16bf a;                                      // A row m = col (an l), strided reads of u
#pragma unroll
    for (int j = 0; j < 8; ++j) {
      int kA = k0 + (khalf << 3) + j;
      a[j]     = (__bf16)ub[(size_t)kA * LL + l0 + col];
      a[8 + j] = (__bf16)ub[(size_t)(kA + 16) * LL + l0 + col];
    }
#pragma unroll
    for (int nt = 0; nt < 4; ++nt) {              // 4 p-tiles (P=64)
      v16bf bf = load_bf_frag(Bp0 + (size_t)(nt * 16 + col) * HH + k0, khalf);
      acc[nt] = __builtin_amdgcn_wmma_f32_16x16x32_bf16(false, a, false, bf,
                                                        (short)0, acc[nt], false, false);
    }
  }
#pragma unroll
  for (int nt = 0; nt < 4; ++nt)
#pragma unroll
    for (int v = 0; v < 8; ++v) {
      int m = v + (khalf << 3);
      Bu[((size_t)b * LL + l0 + m) * PP + nt * 16 + col] = acc[nt][v];
    }
}

// ---------------- scan pass 1: chunk-local scan from zero, emit final state ----------------
__global__ void k_scan_partial(const float* __restrict__ Bu, const float* __restrict__ scal,
                               float* __restrict__ chunkF) {
  const int b = blockIdx.x / CCH, c = blockIdx.x % CCH, p = threadIdx.x;
  const float m11 = scal[p], m12 = scal[PP + p], m21 = scal[2 * PP + p], m22 = scal[3 * PP + p];
  const float s = scal[4 * PP + p];
  const float f1 = m11 * s, f2 = m21 * s;         // b_t = (M11*s*Bu, M21*s*Bu)
  const float* bu = Bu + ((size_t)b * LL + (size_t)c * CLEN) * PP + p;
  float x1 = 0.f, x2 = 0.f;
  for (int t = 0; t < CLEN; ++t) {
    float bv = bu[(size_t)t * PP];
    float n1 = fmaf(m11, x1, fmaf(m12, x2, f1 * bv));
    float n2 = fmaf(m21, x1, fmaf(m22, x2, f2 * bv));
    x1 = n1; x2 = n2;
  }
  size_t idx = (((size_t)b * CCH + c) * PP + p) * 2;
  chunkF[idx] = x1; chunkF[idx + 1] = x2;
}

// ---------------- scan pass 2: sequential carry across chunks via M^CLEN ----------------
__global__ void k_scan_carry(const float* __restrict__ chunkF, const float* __restrict__ scal,
                             float* __restrict__ carry) {
  const int tid = threadIdx.x;                    // BB*PP = 1024
  const int b = tid >> 6, p = tid & 63;
  float a = scal[p], bb = scal[PP + p], cc = scal[2 * PP + p], dd = scal[3 * PP + p];
  for (int i = 0; i < 8; ++i) {                   // M^(2^8) by squaring
    float na = a * a + bb * cc, nb = a * bb + bb * dd;
    float nc = cc * a + dd * cc, nd = cc * bb + dd * dd;
    a = na; bb = nb; cc = nc; dd = nd;
  }
  float e1 = 0.f, e2 = 0.f;
  for (int c = 0; c < CCH; ++c) {
    size_t idx = (((size_t)b * CCH + c) * PP + p) * 2;
    carry[idx] = e1; carry[idx + 1] = e2;         // exclusive carry for chunk c
    float g1 = chunkF[idx], g2 = chunkF[idx + 1];
    float ne1 = a * e1 + bb * e2 + g1;
    float ne2 = cc * e1 + dd * e2 + g2;
    e1 = ne1; e2 = ne2;
  }
}

// ---------------- scan pass 3: re-scan seeded with carry, stream x2 ----------------
__global__ void k_scan_final(const float* __restrict__ Bu, const float* __restrict__ scal,
                             const float* __restrict__ carry, float* __restrict__ X2) {
  const int b = blockIdx.x / CCH, c = blockIdx.x % CCH, p = threadIdx.x;
  const float m11 = scal[p], m12 = scal[PP + p], m21 = scal[2 * PP + p], m22 = scal[3 * PP + p];
  const float s = scal[4 * PP + p];
  const float f1 = m11 * s, f2 = m21 * s;
  size_t cidx = (((size_t)b * CCH + c) * PP + p) * 2;
  float x1 = carry[cidx], x2 = carry[cidx + 1];
  const size_t base = ((size_t)b * LL + (size_t)c * CLEN) * PP + p;
  const float* bu = Bu + base;
  float* xo = X2 + base;
  for (int t = 0; t < CLEN; ++t) {
    float bv = bu[(size_t)t * PP];
    float n1 = fmaf(m11, x1, fmaf(m12, x2, f1 * bv));
    float n2 = fmaf(m21, x1, fmaf(m22, x2, f2 * bv));
    x1 = n1; x2 = n2;
    xo[(size_t)t * PP] = x2;
  }
}

// ---------------- GEMM 2 + D-skip + exact GELU: ys (bf16, row-major (b*L, H)) ----------------
__global__ void __launch_bounds__(256) k_cproj_gelu(const float* __restrict__ X2,
                                                    const __bf16* __restrict__ Cp0,
                                                    const float* __restrict__ u,
                                                    const float* __restrict__ Dv,
                                                    __bf16* __restrict__ ys) {
  const int lane  = threadIdx.x & 31;
  const int wave  = threadIdx.x >> 5;
  const int tile  = blockIdx.x * 8 + wave;
  const int b     = tile / (LL / 16);
  const int l0    = (tile % (LL / 16)) * 16;
  const int col   = lane & 15;
  const int khalf = lane >> 4;

  v16bf a[2];                                     // K = P = 64 -> 2 fragments, preloaded
  const float* xrow = X2 + ((size_t)b * LL + l0 + col) * PP;
#pragma unroll
  for (int ki = 0; ki < 2; ++ki) {
    v16bf f;
#pragma unroll
    for (int j = 0; j < 8; ++j) {
      f[j]     = (__bf16)xrow[ki * 32 + (khalf << 3) + j];
      f[8 + j] = (__bf16)xrow[ki * 32 + 16 + (khalf << 3) + j];
    }
    a[ki] = f;
  }
  const float* ub = u + (size_t)b * HH * LL;
  for (int ht = 0; ht < HH / 16; ++ht) {
    v8f acc = {};
#pragma unroll
    for (int ki = 0; ki < 2; ++ki) {
      v16bf bf = load_bf_frag(Cp0 + (size_t)(ht * 16 + col) * PP + ki * 32, khalf);
      acc = __builtin_amdgcn_wmma_f32_16x16x32_bf16(false, a[ki], false, bf,
                                                    (short)0, acc, false, false);
    }
    const int h = ht * 16 + col;
    const float dh = Dv[h];
#pragma unroll
    for (int v = 0; v < 8; ++v) {
      int m = v + (khalf << 3);
      int l = l0 + m;
      float yv = acc[v] + ub[(size_t)h * LL + l] * dh;
      float g  = 0.5f * yv * (1.0f + erff(yv * 0.70710678118654752f));   // exact GELU
      ys[((size_t)b * LL + l) * HH + h] = (__bf16)g;
    }
  }
}

// ---------------- GEMM 3+4: out = (ys@W1^T + b1) * sigmoid(ys@W2^T + b2) ----------------
// Weight slabs for the current 16 output columns are staged in LDS once per block
// and shared by all 8 waves (8x L2-traffic reduction; B fragments come from ds_load).
__global__ void __launch_bounds__(256) k_mlp_out(const __bf16* __restrict__ ys,
                                                 const __bf16* __restrict__ W1b,
                                                 const __bf16* __restrict__ W2b,
                                                 const float* __restrict__ b1,
                                                 const float* __restrict__ b2,
                                                 float* __restrict__ out) {
  __shared__ __bf16 sW[2][16 * WPAD];             // 2 x 16.25 KB
  const int lane  = threadIdx.x & 31;
  const int wave  = threadIdx.x >> 5;
  const int tile  = blockIdx.x * 8 + wave;        // n-tile over B*L/16
  const int n0    = tile * 16;
  const int col   = lane & 15;
  const int khalf = lane >> 4;

  v16bf a[16];                                    // preload full K=512 A fragments (128 VGPRs)
  const __bf16* arow = ys + (size_t)(n0 + col) * HH;
#pragma unroll
  for (int ki = 0; ki < 16; ++ki) a[ki] = load_bf_frag(arow + ki * 32, khalf);

  for (int ht = 0; ht < HH / 16; ++ht) {
    __syncthreads();
    // cooperative stage of 16 rows x 512 of W1 and W2 (1024 x 16B chunks each)
#pragma unroll
    for (int it = 0; it < 4; ++it) {
      int chunk = threadIdx.x + it * 256;         // 0..1023
      int r  = chunk >> 6;                        // row 0..15
      int cc = (chunk & 63) << 3;                 // col 0,8,...,504
      *(v8bf*)&sW[0][r * WPAD + cc] = *(const v8bf*)(W1b + (size_t)(ht * 16 + r) * HH + cc);
      *(v8bf*)&sW[1][r * WPAD + cc] = *(const v8bf*)(W2b + (size_t)(ht * 16 + r) * HH + cc);
    }
    __syncthreads();

    v8f acc1 = {}, acc2 = {};
    const __bf16* w1row = &sW[0][col * WPAD];
    const __bf16* w2row = &sW[1][col * WPAD];
#pragma unroll 4
    for (int ki = 0; ki < 16; ++ki) {
      v16bf bw1 = load_bf_frag(w1row + ki * 32, khalf);   // LDS, conflict-free via WPAD skew
      v16bf bw2 = load_bf_frag(w2row + ki * 32, khalf);
      acc1 = __builtin_amdgcn_wmma_f32_16x16x32_bf16(false, a[ki], false, bw1,
                                                     (short)0, acc1, false, false);
      acc2 = __builtin_amdgcn_wmma_f32_16x16x32_bf16(false, a[ki], false, bw2,
                                                     (short)0, acc2, false, false);
    }
    const int h = ht * 16 + col;
    const float bias1 = b1[h], bias2 = b2[h];
#pragma unroll
    for (int v = 0; v < 8; ++v) {
      int m    = v + (khalf << 3);
      int nrow = n0 + m;
      int b    = nrow >> 12;                      // / LL
      int l    = nrow & (LL - 1);
      float pre  = acc1[v] + bias1;
      float gate = 1.0f / (1.0f + expf(-(acc2[v] + bias2)));
      out[((size_t)b * HH + h) * LL + l] = pre * gate;   // (B,H,L) layout
    }
  }
}

extern "C" void kernel_launch(void* const* d_in, const int* in_sizes, int n_in,
                              void* d_out, int out_size, void* d_ws, size_t ws_size,
                              hipStream_t stream) {
  const float* u      = (const float*)d_in[0];
  const float* A_diag = (const float*)d_in[1];
  const float* Bp     = (const float*)d_in[2];   // (P,H,2)
  const float* Cp     = (const float*)d_in[3];   // (H,P,2)
  const float* Dv     = (const float*)d_in[4];
  const float* steps  = (const float*)d_in[5];
  const float* W1     = (const float*)d_in[6];
  const float* b1     = (const float*)d_in[7];
  const float* W2     = (const float*)d_in[8];
  const float* b2     = (const float*)d_in[9];
  float* out = (float*)d_out;

  char* p = (char*)d_ws;
  auto carve = [&](size_t bytes) { char* r = p; p += (bytes + 255) & ~(size_t)255; return (void*)r; };
  float*  scal   = (float*) carve(5 * PP * sizeof(float));
  __bf16* Bp0    = (__bf16*)carve((size_t)PP * HH * sizeof(__bf16));
  __bf16* Cp0    = (__bf16*)carve((size_t)HH * PP * sizeof(__bf16));
  __bf16* W1b    = (__bf16*)carve((size_t)HH * HH * sizeof(__bf16));
  __bf16* W2b    = (__bf16*)carve((size_t)HH * HH * sizeof(__bf16));
  float*  Bu     = (float*) carve((size_t)BB * LL * PP * sizeof(float));   // 16.8 MB
  float*  X2     = (float*) carve((size_t)BB * LL * PP * sizeof(float));   // 16.8 MB
  float*  chunkF = (float*) carve((size_t)BB * CCH * PP * 2 * sizeof(float));
  float*  carry  = (float*) carve((size_t)BB * CCH * PP * 2 * sizeof(float));
  __bf16* ysbuf  = (__bf16*)carve((size_t)BB * LL * HH * sizeof(__bf16));  // 67 MB (L2-resident)

  k_prep<<<1, PP, 0, stream>>>(A_diag, steps, scal);
  k_cvt<<<(PP * HH + 255) / 256, 256, 0, stream>>>(Bp, Bp0, PP * HH, 2);
  k_cvt<<<(HH * PP + 255) / 256, 256, 0, stream>>>(Cp, Cp0, HH * PP, 2);
  k_cvt<<<(HH * HH + 255) / 256, 256, 0, stream>>>(W1, W1b, HH * HH, 1);
  k_cvt<<<(HH * HH + 255) / 256, 256, 0, stream>>>(W2, W2b, HH * HH, 1);

  k_gemm_bu<<<(BB * LL / 16) / 8, 256, 0, stream>>>(u, Bp0, Bu);
  k_scan_partial<<<BB * CCH, PP, 0, stream>>>(Bu, scal, chunkF);
  k_scan_carry<<<1, BB * PP, 0, stream>>>(chunkF, scal, carry);
  k_scan_final<<<BB * CCH, PP, 0, stream>>>(Bu, scal, carry, X2);
  k_cproj_gelu<<<(BB * LL / 16) / 8, 256, 0, stream>>>(X2, Cp0, u, Dv, ysbuf);
  k_mlp_out<<<(BB * LL / 16) / 8, 256, 0, stream>>>(ysbuf, W1b, W2b, b1, b2, out);
}
